// MultiHeadAttentionBlock_swin_37890201485780
// MI455X (gfx1250) — matrix operand order repaired and checked by reading, not compiled, not run on hardware
//
#include <hip/hip_runtime.h>
#include <math.h>

#define T_DIM 64
#define S_DIM 64
#define D_DIM 512
#define H_NUM 16
#define HD    32
#define NTHREADS 256
#define NWAVES 8

typedef __attribute__((ext_vector_type(16))) __bf16 v16bf;
typedef __attribute__((ext_vector_type(8)))  __bf16 v8bf;
typedef __attribute__((ext_vector_type(4)))  __bf16 v4bf;
typedef __attribute__((ext_vector_type(8)))  float  v8f;

__device__ __forceinline__ v8f wmma_bf16(v16bf a, v16bf b, v8f c) {
  return __builtin_amdgcn_wmma_f32_16x16x32_bf16(false, a, false, b, (short)0, c,
                                                 false, false);
}

// A-frag (16x32 bf16, M x K): lane holds row m = m0 + (lane&15);
// K chunks: [k0 + half*8 .. +7] and [k0 + 16 + half*8 .. +7]  (half = lane>>4)
__device__ __forceinline__ v16bf load_a_frag(const __bf16* base, int ld, int m0, int k0) {
  const int lane = threadIdx.x & 31;
  const int half = lane >> 4;
  const __bf16* p = base + (m0 + (lane & 15)) * ld + k0 + half * 8;
  v8bf c0 = *(const v8bf*)p;
  v8bf c1 = *(const v8bf*)(p + 16);
  v16bf r;
#pragma unroll
  for (int i = 0; i < 8; ++i) { r[i] = c0[i]; r[8 + i] = c1[i]; }
  return r;
}

// B-frag (32x16 bf16, K x N) loaded from an N-major buffer Bt[n][k]:
// lane holds column n = n0 + (lane&15), K chunks: [k0 + half*8 .. +7] and [k0 + 16 + half*8 .. +7]
__device__ __forceinline__ v16bf load_b_frag(const __bf16* base, int ld, int n0, int k0) {
  const int lane = threadIdx.x & 31;
  const int half = lane >> 4;
  const __bf16* p = base + (n0 + (lane & 15)) * ld + k0 + half * 8;
  v8bf c0 = *(const v8bf*)p;
  v8bf c1 = *(const v8bf*)(p + 16);
  v16bf r;
#pragma unroll
  for (int i = 0; i < 8; ++i) { r[i] = c0[i]; r[8 + i] = c1[i]; }
  return r;
}

// Low 32 bits of a generic shared pointer = LDS offset (flat-LDS aperture rule)
__device__ __forceinline__ unsigned lds_off(const void* p) {
  return (unsigned)(unsigned long long)p;
}

// CDNA5 async global->LDS copy (ASYNCcnt-tracked, §15.18.3 opcode 98)
__device__ __forceinline__ void async_g2l_b128(unsigned ldsoff, const void* g) {
  asm volatile("global_load_async_to_lds_b128 %0, %1, off"
               :: "v"(ldsoff), "v"(g) : "memory");
}
__device__ __forceinline__ void wait_async() {
  asm volatile("s_wait_asynccnt 0x0" ::: "memory");
}

// GEMM: Out(64x512) = A(64x512, LDS bf16) @ W^T  (W row-major = N-major B operand)
// 2x2 register blocking: 4 independent accumulator chains, 2 A-frags + 2 B-frags
// per k-step feed 4 WMMAs (1.0 frag-loads/WMMA).
// MODE 0: bf16 -> dstL[t][col] (ld=512) | MODE 1: bf16 -> dstL[col][t] (ld=64,
// packed b128 stores) | MODE 2: f32 -> dstG[t][col] (global)
template <int MODE>
__device__ __forceinline__ void proj_gemm(const __bf16* __restrict__ A,
                                          const __bf16* __restrict__ W,
                                          __bf16* dstL, float* dstG) {
  const int wave = threadIdx.x >> 5;
  const int lane = threadIdx.x & 31;
  const int n = lane & 15, half = lane >> 4;
  for (int it = wave; it < 32; it += NWAVES) {
    const int mp = it >> 4;   // 0..1  -> m-tiles 2*mp, 2*mp+1 (32 rows)
    const int np = it & 15;   // 0..15 -> cols np*32 .. np*32+31
    v8f acc[2][2];
#pragma unroll
    for (int mi = 0; mi < 2; ++mi)
#pragma unroll
      for (int ni = 0; ni < 2; ++ni)
#pragma unroll
        for (int i = 0; i < 8; ++i) acc[mi][ni][i] = 0.f;
#pragma unroll 2
    for (int k = 0; k < 16; ++k) {
      v16bf a0 = load_a_frag(A, D_DIM, (2 * mp) * 16,     k * 32);
      v16bf a1 = load_a_frag(A, D_DIM, (2 * mp + 1) * 16, k * 32);
      v16bf b0 = load_b_frag(W, D_DIM, np * 32,      k * 32);
      v16bf b1 = load_b_frag(W, D_DIM, np * 32 + 16, k * 32);
      acc[0][0] = wmma_bf16(a0, b0, acc[0][0]);
      acc[0][1] = wmma_bf16(a0, b1, acc[0][1]);
      acc[1][0] = wmma_bf16(a1, b0, acc[1][0]);
      acc[1][1] = wmma_bf16(a1, b1, acc[1][1]);
    }
#pragma unroll
    for (int mi = 0; mi < 2; ++mi) {
      const int t0 = (2 * mp + mi) * 16 + 8 * half;
#pragma unroll
      for (int ni = 0; ni < 2; ++ni) {
        const int col = np * 32 + ni * 16 + n;
        if (MODE == 1) {
          v8bf pk;
#pragma unroll
          for (int i = 0; i < 8; ++i) pk[i] = (__bf16)acc[mi][ni][i];
          *(v8bf*)(dstL + col * T_DIM + t0) = pk;   // one ds_store_b128
        } else if (MODE == 0) {
#pragma unroll
          for (int i = 0; i < 8; ++i)
            dstL[(t0 + i) * D_DIM + col] = (__bf16)acc[mi][ni][i];
        } else {
#pragma unroll
          for (int i = 0; i < 8; ++i)
            dstG[(size_t)(t0 + i) * D_DIM + col] = acc[mi][ni][i];
        }
      }
    }
  }
}

__global__ void convert_w_kernel(const float* __restrict__ Wq, const float* __restrict__ Wk,
                                 const float* __restrict__ Wv, const float* __restrict__ Wo,
                                 unsigned short* __restrict__ dst_raw) {
  __bf16* dst = (__bf16*)dst_raw;
  const int per = D_DIM * D_DIM;
  int idx = blockIdx.x * blockDim.x + threadIdx.x;
  int m = idx / per;
  int off = idx - m * per;
  const float* src = (m == 0) ? Wq : (m == 1) ? Wk : (m == 2) ? Wv : Wo;
  dst[idx] = (__bf16)src[off];
}

__global__ __launch_bounds__(NTHREADS)
void swin_attn_kernel(const float* __restrict__ qg, const float* __restrict__ kg,
                      const float* __restrict__ vg, const float* __restrict__ bias,
                      const unsigned char* __restrict__ mask,
                      const unsigned short* __restrict__ Wraw,
                      const float* __restrict__ logit_scale,
                      float* __restrict__ out) {
  extern __shared__ char smem[];
  __bf16* xb     = (__bf16*)smem;              // [0,64K): bf16 x tile
  float*  xstage = (float*)(smem + 65536);     // [64K,192K): async fp32 staging
  __bf16* Vt     = (__bf16*)(smem + 65536);    // [64K,128K): V^T, after v consumed
  __bf16* Pbase  = (__bf16*)(smem + 131072);   // [128K,144K): per-wave P tiles
  __bf16* Qn     = (__bf16*)(smem + 196608);   // [192K,256K): Qnorm, later vals
  __bf16* Kn     = (__bf16*)(smem + 262144);   // [256K,320K): Knorm
  const __bf16* Wbf = (const __bf16*)Wraw;

  const int b = blockIdx.x;
  const int tid = threadIdx.x;
  const int wave = tid >> 5;
  const int lane = tid & 31;
  const int nn = lane & 15, half = lane >> 4;

  // Issue async HBM->LDS copy of one 128KB x tile (32 b128 issues per thread)
  auto stage_async = [&](const float* src) {
    const char* g = (const char*)(src + (size_t)b * T_DIM * D_DIM);
    const unsigned l0 = lds_off(xstage);
    for (int i = tid; i < (T_DIM * D_DIM * 4) / 16; i += NTHREADS)
      async_g2l_b128(l0 + i * 16, g + (size_t)i * 16);
  };
  // fp32 staging -> bf16 xb
  auto convert_x = [&]() {
    const float4* s4 = (const float4*)xstage;
    for (int i = tid; i < T_DIM * D_DIM / 4; i += NTHREADS) {
      float4 f = s4[i];
      v4bf o;
      o[0] = (__bf16)f.x; o[1] = (__bf16)f.y; o[2] = (__bf16)f.z; o[3] = (__bf16)f.w;
      *((v4bf*)xb + i) = o;
    }
  };
  auto normalize = [&](__bf16* M, bool useScale) {
    for (int idx = tid; idx < T_DIM * H_NUM; idx += NTHREADS) {
      int t = idx >> 4, h = idx & 15;
      __bf16* p = M + t * D_DIM + h * HD;
      float ss = 0.f;
#pragma unroll
      for (int j = 0; j < HD; ++j) { float x = (float)p[j]; ss += x * x; }
      float inv = 1.f / fmaxf(sqrtf(ss), 1e-12f);
      if (useScale) inv *= __expf(fminf(logit_scale[h], 4.6051702f)); // log(100)
#pragma unroll
      for (int j = 0; j < HD; ++j) p[j] = (__bf16)((float)p[j] * inv);
    }
  };

  // ---- Stage 1: projections, async-pipelined staging ----
  stage_async(qg);
  wait_async(); __syncthreads();
  convert_x(); __syncthreads();                    // xb = bf16(q); xstage dead

  stage_async(kg);                                 // prefetch k under Q GEMM
  proj_gemm<0>(xb, Wbf + 0 * D_DIM * D_DIM, Qn, nullptr);
  __syncthreads();
  normalize(Qn, true);
  wait_async(); __syncthreads();
  convert_x(); __syncthreads();                    // xb = bf16(k)

  stage_async(vg);                                 // prefetch v under K GEMM
  proj_gemm<0>(xb, Wbf + 1 * D_DIM * D_DIM, Kn, nullptr);
  __syncthreads();
  normalize(Kn, false);
  wait_async(); __syncthreads();
  convert_x(); __syncthreads();                    // xb = bf16(v)

  proj_gemm<1>(xb, Wbf + 2 * D_DIM * D_DIM, Vt, nullptr);  // Vt overlays dead staging
  __syncthreads();

  // ---- Stage 2: attention (each wave owns heads {wave, wave+8}) ----
  __bf16* Pw = Pbase + wave * (16 * S_DIM);  // 2KB wave-private P tile
  for (int h = wave; h < H_NUM; h += NWAVES) {
    for (int mt = 0; mt < 4; ++mt) {
      v16bf aq = load_a_frag(Qn, D_DIM, mt * 16, h * HD);
      v8f acc[4];
#pragma unroll
      for (int st = 0; st < 4; ++st) {
        v16bf bk = load_b_frag(Kn, D_DIM, st * 16, h * HD);
        v8f z;
#pragma unroll
        for (int i = 0; i < 8; ++i) z[i] = 0.f;
        acc[st] = wmma_bf16(aq, bk, z);
      }
      // bias + mask; element (t,s): t = mt*16 + i + 8*half, s = st*16 + nn
#pragma unroll
      for (int st = 0; st < 4; ++st) {
        const int s = st * 16 + nn;
#pragma unroll
        for (int i = 0; i < 8; ++i) {
          const int t = mt * 16 + i + 8 * half;
          float bv = bias[(t * S_DIM + s) * H_NUM + h];
          bool mk = mask[((size_t)b * T_DIM + t) * S_DIM + s] != 0;
          acc[st][i] = mk ? -1e10f : (acc[st][i] + bv);
        }
      }
      // softmax over s: row lives in 16 lanes (same half) x 4 fragments
#pragma unroll
      for (int i = 0; i < 8; ++i) {
        float m = fmaxf(fmaxf(acc[0][i], acc[1][i]), fmaxf(acc[2][i], acc[3][i]));
#pragma unroll
        for (int off = 1; off < 16; off <<= 1) m = fmaxf(m, __shfl_xor(m, off, 32));
        float sum = 0.f;
#pragma unroll
        for (int st = 0; st < 4; ++st) { acc[st][i] = __expf(acc[st][i] - m); sum += acc[st][i]; }
#pragma unroll
        for (int off = 1; off < 16; off <<= 1) sum += __shfl_xor(sum, off, 32);
        float inv = 1.f / sum;
#pragma unroll
        for (int st = 0; st < 4; ++st) acc[st][i] *= inv;
      }
      // reshape D-frag -> A-frag through wave-private LDS tile
#pragma unroll
      for (int st = 0; st < 4; ++st)
#pragma unroll
        for (int i = 0; i < 8; ++i)
          Pw[(i + 8 * half) * S_DIM + st * 16 + nn] = (__bf16)acc[st][i];
      v16bf ap0 = load_a_frag(Pw, S_DIM, 0, 0);
      v16bf ap1 = load_a_frag(Pw, S_DIM, 0, 32);
      // P(16x64) @ V(64x32): Vt is N-major [d][s]
#pragma unroll
      for (int dt = 0; dt < 2; ++dt) {
        v8f vac;
#pragma unroll
        for (int i = 0; i < 8; ++i) vac[i] = 0.f;
        vac = wmma_bf16(ap0, load_b_frag(Vt, T_DIM, h * HD + dt * 16, 0), vac);
        vac = wmma_bf16(ap1, load_b_frag(Vt, T_DIM, h * HD + dt * 16, 32), vac);
        const int col = h * HD + dt * 16 + nn;
#pragma unroll
        for (int i = 0; i < 8; ++i) {
          const int t = mt * 16 + i + 8 * half;
          Qn[t * D_DIM + col] = (__bf16)vac[i];   // vals overwrite this head's Q cols
        }
      }
    }
  }
  __syncthreads();

  // ---- Stage 3: output projection ----
  proj_gemm<2>(Qn, Wbf + 3 * D_DIM * D_DIM, nullptr, out + (size_t)b * T_DIM * D_DIM);
}

extern "C" void kernel_launch(void* const* d_in, const int* in_sizes, int n_in,
                              void* d_out, int out_size, void* d_ws, size_t ws_size,
                              hipStream_t stream) {
  const float* q  = (const float*)d_in[0];
  const float* k  = (const float*)d_in[1];
  const float* v  = (const float*)d_in[2];
  const float* bias = (const float*)d_in[3];
  const unsigned char* mask = (const unsigned char*)d_in[4];
  const float* Wq = (const float*)d_in[5];
  const float* Wk = (const float*)d_in[6];
  const float* Wv = (const float*)d_in[7];
  const float* Wo = (const float*)d_in[8];
  const float* logit_scale = (const float*)d_in[9];
  float* out = (float*)d_out;
  unsigned short* Wbf = (unsigned short*)d_ws;  // 4 x 512 x 512 bf16 = 2 MB

  const int B = in_sizes[0] / (T_DIM * D_DIM);

  convert_w_kernel<<<(4 * D_DIM * D_DIM) / NTHREADS, NTHREADS, 0, stream>>>(
      Wq, Wk, Wv, Wo, Wbf);

  const size_t shmem = 5 * 65536;  // 320 KB: full WGP LDS
  swin_attn_kernel<<<B, NTHREADS, shmem, stream>>>(
      q, k, v, bias, mask, Wbf, logit_scale, out);
}
